// GCN_spatial_32512902431511
// MI455X (gfx1250) — matrix operand order repaired and checked by reading, not compiled, not run on hardware
//
#include <hip/hip_runtime.h>
#include <hip/hip_bf16.h>

typedef __bf16 bf16_t;
typedef __attribute__((ext_vector_type(16))) __bf16 v16bf;
typedef __attribute__((ext_vector_type(8)))  __bf16 v8bf;
typedef __attribute__((ext_vector_type(4)))  __bf16 v4bf;
typedef __attribute__((ext_vector_type(8)))  float  v8f;
typedef __attribute__((ext_vector_type(4)))  float  v4f;

#define RT 2  // 16-row tiles per workgroup (32 rows): 313 WGs, rows disjoint ->
              // adj streamed from HBM exactly once per layer

// ---------------------------------------------------------------------------
// x [B=8,16,N] f32  ->  Ht [c=b*16+f][n] bf16   (flat layouts are identical)
// ---------------------------------------------------------------------------
__global__ __launch_bounds__(256) void cvt_in_kernel(
    const float* __restrict__ x, bf16_t* __restrict__ Ht, int total) {
  int idx = blockIdx.x * blockDim.x + threadIdx.x;
  if (idx < total) Ht[idx] = (bf16_t)x[idx];
}

// ---------------------------------------------------------------------------
// adj f32 [N*N] -> bf16 [N*N], 4 elements per thread (N*N % 4 == 0)
// ---------------------------------------------------------------------------
__global__ __launch_bounds__(256) void cvt_adj_kernel(
    const float* __restrict__ A, bf16_t* __restrict__ Abf, int total4) {
  int idx = blockIdx.x * blockDim.x + threadIdx.x;
  if (idx >= total4) return;
  const v4f a = *(const v4f*)(A + (size_t)idx * 4);
  v4bf o;
#pragma unroll
  for (int i = 0; i < 4; ++i) o[i] = (bf16_t)a[i];
  *(v4bf*)(Abf + (size_t)idx * 4) = o;
}

// ---------------------------------------------------------------------------
// AGG[N][C] (f32) = A[N][N] x Ht[C][N] (bf16)
// A is either pre-converted bf16 (ABF16=1: pure loads, 2x16B per fragment) or
// f32 converted in registers (ABF16=0 fallback when workspace is small).
// Workgroup: 32 rows x C cols, 8 waves; wave owns RT row-tiles x NSETS
// col-tiles; NSETS = C / (16 * 8).
// Per k-chunk: load all B frags, build all A frags, then burst RT*NSETS WMMAs
// (accumulator chaining has no A/B hazard; the WMMA->VALU WAR window is
// covered by the next iteration's loads).
// Fragment layouts per CDNA5 ISA 7.12.2 (wave32):
//   A 16x32 bf16 : lane r(=lane&15)=row; half=lane>>4:
//                  elems 0..7 -> K=8*half+0..7 ; elems 8..15 -> K=16+8*half+0..7
//   B 32x16 bf16 : lane r = col; K as A: 8*half+0..7, 16+8*half+0..7 (two
//                  16B loads from column-major Ht)
//   C/D 16x16 f32: elem v -> row v+8*half ; lane r -> col
// ---------------------------------------------------------------------------
template <int NSETS, bool ABF16>
__global__ __launch_bounds__(256) void agg_kernel(
    const void* __restrict__ Av, const bf16_t* __restrict__ Ht,
    float* __restrict__ AGG, int N, int C) {
  const int lane = threadIdx.x & 31;
  const int wave = threadIdx.x >> 5;
  const int r    = lane & 15;
  const int half = lane >> 4;
  const int rowBase = blockIdx.x * (16 * RT);

  // Per-row-tile A pointers (pre-offset by the 8*half K sub-block)
  const float*  aRowF[RT];
  const bf16_t* aRowB[RT];
#pragma unroll
  for (int t = 0; t < RT; ++t) {
    int row = rowBase + 16 * t + r;
    if (row >= N) row = N - 1;  // clamp; stores are guarded
    aRowF[t] = (const float*)Av  + (size_t)row * N + 8 * half;
    aRowB[t] = (const bf16_t*)Av + (size_t)row * N + 8 * half;
  }

  // Per-col-tile B pointers (column-major H, pre-offset by the 8*half K sub-block)
  const bf16_t* bPtr[NSETS];
#pragma unroll
  for (int s = 0; s < NSETS; ++s) {
    const int col = (wave * NSETS + s) * 16 + r;
    bPtr[s] = Ht + (size_t)col * N + 8 * half;
  }

  const v8f vzero = {0.f, 0.f, 0.f, 0.f, 0.f, 0.f, 0.f, 0.f};
  v8f acc[RT][NSETS];
#pragma unroll
  for (int t = 0; t < RT; ++t)
#pragma unroll
    for (int s = 0; s < NSETS; ++s) acc[t][s] = vzero;

  const int kMain = N & ~31;
  for (int k0 = 0; k0 < kMain; k0 += 32) {
    // ---- stage 1: all B fragments (2x16B loads from L2-resident Ht: K 8h+0..7, 16+8h+0..7)
    v16bf bfrag[NSETS];
#pragma unroll
    for (int s = 0; s < NSETS; ++s) {
      const v8bf lo = *(const v8bf*)(bPtr[s] + k0);
      const v8bf hi = *(const v8bf*)(bPtr[s] + k0 + 16);
      bfrag[s] = __builtin_shufflevector(lo, hi, 0, 1, 2, 3, 4, 5, 6, 7,
                                         8, 9, 10, 11, 12, 13, 14, 15);
    }

    // ---- stage 2: all A fragments
    v16bf afrag[RT];
#pragma unroll
    for (int t = 0; t < RT; ++t) {
      if (ABF16) {
        const v8bf lo = *(const v8bf*)(aRowB[t] + k0);        // K 8h+0..7
        const v8bf hi = *(const v8bf*)(aRowB[t] + k0 + 16);   // K 16+8h+0..7
        afrag[t] = __builtin_shufflevector(lo, hi, 0, 1, 2, 3, 4, 5, 6, 7,
                                           8, 9, 10, 11, 12, 13, 14, 15);
      } else {
        const v4f a0 = *(const v4f*)(aRowF[t] + k0);
        const v4f a1 = *(const v4f*)(aRowF[t] + k0 + 4);
        const v4f a2 = *(const v4f*)(aRowF[t] + k0 + 16);
        const v4f a3 = *(const v4f*)(aRowF[t] + k0 + 20);
#pragma unroll
        for (int i = 0; i < 4; ++i) {
          afrag[t][i]      = (bf16_t)a0[i];
          afrag[t][4 + i]  = (bf16_t)a1[i];
          afrag[t][8 + i]  = (bf16_t)a2[i];
          afrag[t][12 + i] = (bf16_t)a3[i];
        }
      }
    }

    // ---- stage 3: WMMA burst
#pragma unroll
    for (int t = 0; t < RT; ++t)
#pragma unroll
      for (int s = 0; s < NSETS; ++s)
        acc[t][s] = __builtin_amdgcn_wmma_f32_16x16x32_bf16(
            false, afrag[t], false, bfrag[s], (short)0, acc[t][s], false, false);
  }

  // K tail (N % 32 == 16 here): zero-pad per lane; runs once per workgroup
  if (kMain < N) {
    const int k0 = kMain;
    v16bf bfrag[NSETS];
#pragma unroll
    for (int s = 0; s < NSETS; ++s) {
#pragma unroll
      for (int i = 0; i < 16; ++i) {
        const int off = (i < 8) ? i : (i + 8);
        const int kk = k0 + 8 * half + off;
        bfrag[s][i] = (kk < N) ? bPtr[s][k0 + off] : (bf16_t)0.f;
      }
    }
    v16bf afrag[RT];
#pragma unroll
    for (int t = 0; t < RT; ++t) {
#pragma unroll
      for (int i = 0; i < 16; ++i) {
        const int off = (i < 8) ? i : (i + 8);  // local K: 0..7, 16..23
        const int kk  = k0 + 8 * half + off;
        if (ABF16) afrag[t][i] = (kk < N) ? aRowB[t][k0 + off] : (bf16_t)0.f;
        else       afrag[t][i] = (kk < N) ? (bf16_t)aRowF[t][k0 + off] : (bf16_t)0.f;
      }
    }
#pragma unroll
    for (int t = 0; t < RT; ++t)
#pragma unroll
      for (int s = 0; s < NSETS; ++s)
        acc[t][s] = __builtin_amdgcn_wmma_f32_16x16x32_bf16(
            false, afrag[t], false, bfrag[s], (short)0, acc[t][s], false, false);
  }

  // Store C/D fragments (coalesced across lanes)
#pragma unroll
  for (int t = 0; t < RT; ++t) {
#pragma unroll
    for (int s = 0; s < NSETS; ++s) {
      const int col = (wave * NSETS + s) * 16 + r;
#pragma unroll
      for (int v = 0; v < 8; ++v) {
        const int row = rowBase + 16 * t + v + 8 * half;
        if (row < N) AGG[(size_t)row * C + col] = acc[t][s][v];
      }
    }
  }
}

// ---------------------------------------------------------------------------
// H'[c=b*Fo+fo][n] = sum_fi AGG[n][b*Fin+fi] * W[fi][fo] + bias[fo]
// bf16 out into Ht (inter-layer), or f32 out straight into d_out (last layer).
// ---------------------------------------------------------------------------
template <bool F32OUT>
__global__ __launch_bounds__(256) void proj_kernel(
    const float* __restrict__ AGG, const float* __restrict__ W,
    const float* __restrict__ bias, bf16_t* __restrict__ HtOut,
    float* __restrict__ Out32, int N, int Fin, int Fo) {
  const int Cout  = 8 * Fo;
  const int total = Cout * N;
  int idx = blockIdx.x * blockDim.x + threadIdx.x;
  if (idx >= total) return;
  const int c  = idx / N;       // consecutive threads -> consecutive n (coalesced writes)
  const int n  = idx - c * N;
  const int b  = c / Fo;
  const int fo = c - b * Fo;
  const float* ag = AGG + (size_t)n * (8 * Fin) + b * Fin;
  float acc = bias[fo];
  for (int fi = 0; fi < Fin; ++fi) acc = fmaf(ag[fi], W[fi * Fo + fo], acc);
  if (F32OUT) Out32[(size_t)c * N + n] = acc;
  else        HtOut[(size_t)c * N + n] = (bf16_t)acc;
}

// ---------------------------------------------------------------------------
extern "C" void kernel_launch(void* const* d_in, const int* in_sizes, int n_in,
                              void* d_out, int out_size, void* d_ws, size_t ws_size,
                              hipStream_t stream) {
  const float* x   = (const float*)d_in[0];
  const float* adj = (const float*)d_in[1];
  const float* W1  = (const float*)d_in[2];
  const float* b1  = (const float*)d_in[3];
  const float* W2  = (const float*)d_in[4];
  const float* b2  = (const float*)d_in[5];
  const float* W3  = (const float*)d_in[6];
  const float* b3  = (const float*)d_in[7];
  const float* W4  = (const float*)d_in[8];
  const float* b4  = (const float*)d_in[9];
  float* out = (float*)d_out;

  const int N = in_sizes[0] / 128;  // B*IN_DIM = 128 -> N = 10000

  // Workspace: AGG f32 [N x 512] | Ht bf16 [512 x N] | (optional) Abf bf16 [N x N]
  const size_t aggBytes = (size_t)N * 512 * sizeof(float);
  const size_t htBytes  = (size_t)N * 512 * sizeof(bf16_t);
  float*  AGG = (float*)d_ws;
  bf16_t* Ht  = (bf16_t*)((char*)d_ws + aggBytes);
  bf16_t* Abf = (bf16_t*)((char*)d_ws + aggBytes + htBytes);
  const bool preconv =
      ws_size >= aggBytes + htBytes + (size_t)N * N * sizeof(bf16_t);

  const int rowBlocks = (N + 16 * RT - 1) / (16 * RT);  // 313

  // H0 = bf16(x), column-major [128][N]
  cvt_in_kernel<<<(128 * N + 255) / 256, 256, 0, stream>>>(x, Ht, 128 * N);

  if (preconv) {  // one-time adj f32 -> bf16 (halves A-side HBM per layer)
    const int total4 = (N * N) / 4;
    cvt_adj_kernel<<<(total4 + 255) / 256, 256, 0, stream>>>(adj, Abf, total4);

    agg_kernel<1, true><<<rowBlocks, 256, 0, stream>>>(Abf, Ht, AGG, N, 128);
    proj_kernel<false><<<(8 * 32 * N + 255) / 256, 256, 0, stream>>>(
        AGG, W1, b1, Ht, nullptr, N, 16, 32);

    agg_kernel<2, true><<<rowBlocks, 256, 0, stream>>>(Abf, Ht, AGG, N, 256);
    proj_kernel<false><<<(8 * 64 * N + 255) / 256, 256, 0, stream>>>(
        AGG, W2, b2, Ht, nullptr, N, 32, 64);

    agg_kernel<4, true><<<rowBlocks, 256, 0, stream>>>(Abf, Ht, AGG, N, 512);
    proj_kernel<false><<<(8 * 32 * N + 255) / 256, 256, 0, stream>>>(
        AGG, W3, b3, Ht, nullptr, N, 64, 32);

    agg_kernel<2, true><<<rowBlocks, 256, 0, stream>>>(Abf, Ht, AGG, N, 256);
    proj_kernel<true><<<(8 * 16 * N + 255) / 256, 256, 0, stream>>>(
        AGG, W4, b4, nullptr, out, N, 32, 16);
  } else {  // fallback: convert adj tiles in registers every layer
    agg_kernel<1, false><<<rowBlocks, 256, 0, stream>>>(adj, Ht, AGG, N, 128);
    proj_kernel<false><<<(8 * 32 * N + 255) / 256, 256, 0, stream>>>(
        AGG, W1, b1, Ht, nullptr, N, 16, 32);

    agg_kernel<2, false><<<rowBlocks, 256, 0, stream>>>(adj, Ht, AGG, N, 256);
    proj_kernel<false><<<(8 * 64 * N + 255) / 256, 256, 0, stream>>>(
        AGG, W2, b2, Ht, nullptr, N, 32, 64);

    agg_kernel<4, false><<<rowBlocks, 256, 0, stream>>>(adj, Ht, AGG, N, 512);
    proj_kernel<false><<<(8 * 32 * N + 255) / 256, 256, 0, stream>>>(
        AGG, W3, b3, Ht, nullptr, N, 64, 32);

    agg_kernel<2, false><<<rowBlocks, 256, 0, stream>>>(adj, Ht, AGG, N, 256);
    proj_kernel<true><<<(8 * 16 * N + 255) / 256, 256, 0, stream>>>(
        AGG, W4, b4, nullptr, out, N, 32, 16);
  }
}
